// HAB_21784074125864
// MI455X (gfx1250) — compile-verified, matrix operand order checked by reading, not run on hardware
//
#include <hip/hip_runtime.h>
#include <hip/hip_bf16.h>

typedef unsigned short u16;
typedef __attribute__((ext_vector_type(16))) __bf16 v16bf;
typedef __attribute__((ext_vector_type(8)))  float  v8f;
typedef __attribute__((ext_vector_type(8)))  unsigned short v8u16;

#define MTOK 65536   // B*H*W tokens
#define HH   128
#define WW   128

__device__ __forceinline__ float bf2f(u16 h) {
  union { unsigned u; float f; } x; x.u = ((unsigned)h) << 16; return x.f;
}
__device__ __forceinline__ u16 f2bf(float f) {
  union { float f; unsigned u; } x; x.f = f;
  unsigned r = x.u + 0x7FFFu + ((x.u >> 16) & 1u);
  return (u16)(r >> 16);
}

union Frag { v16bf v; v8u16 h[2]; };

__device__ __forceinline__ v16bf ldfrag(const u16* base, int ld, int lane) {
  // A/B 16-bit fragment: lane half selects K chunks {k0..k0+7, k0+16..k0+23}
  int r = lane & 15, k0 = (lane >> 4) * 8;
  const u16* p = base + (size_t)r * ld + k0;
  Frag f;
  f.h[0] = *(const v8u16*)p;
  f.h[1] = *(const v8u16*)(p + 16);
  return f.v;
}

__device__ __forceinline__ v8f wmma32(v16bf a, v16bf b, v8f c) {
  return __builtin_amdgcn_wmma_f32_16x16x32_bf16(false, a, false, b, (short)0, c,
                                                 false, false);
}

// ---------------- weight fp32 -> padded bf16 [Np][Kp] ----------------
__global__ void cvt_w_kernel(const float* __restrict__ w, u16* __restrict__ o,
                             int N, int K, int Np, int Kp) {
  int idx = blockIdx.x * 256 + threadIdx.x;
  if (idx >= Np * Kp) return;
  int n = idx / Kp, k = idx - n * Kp;
  o[idx] = (n < N && k < K) ? f2bf(w[(size_t)n * K + k]) : (u16)0;
}

// ---------------- relative-position bias [6][256][256] ----------------
__global__ void build_bias_kernel(const int* __restrict__ rpi,
                                  const float* __restrict__ tbl,
                                  float* __restrict__ bias) {
  int idx = blockIdx.x * 256 + threadIdx.x;
  if (idx >= 6 * 65536) return;
  int h = idx / 65536, nm = idx - h * 65536;
  bias[idx] = tbl[rpi[nm] * 6 + h];
}

// ---------------- layernorm (C=180), wave-per-token ----------------
__global__ __launch_bounds__(256)
void ln_kernel(const float* __restrict__ x, const float* __restrict__ g,
               const float* __restrict__ b, u16* __restrict__ outN,
               u16* __restrict__ outWin) {
  int wv = threadIdx.x >> 5, lane = threadIdx.x & 31;
  size_t t = (size_t)blockIdx.x * 8 + wv;
  const float* xp = x + t * 180;
  float v[6], s = 0.f, sq = 0.f;
  #pragma unroll
  for (int j = 0; j < 6; j++) {
    int c = lane + 32 * j;
    float f = (c < 180) ? xp[c] : 0.f;
    v[j] = f; s += f; sq += f * f;
  }
  #pragma unroll
  for (int m = 16; m >= 1; m >>= 1) {
    s += __shfl_xor(s, m, 32); sq += __shfl_xor(sq, m, 32);
  }
  float mean = s * (1.f / 180.f);
  float var = sq * (1.f / 180.f) - mean * mean;
  float inv = rsqrtf(var + 1e-5f);
  size_t winoff = 0;
  if (outWin) {
    int bb = (int)(t >> 14), rem = (int)(t & 16383);
    int y = rem >> 7, xc = rem & 127;
    int win = (bb * 8 + (y >> 4)) * 8 + (xc >> 4);
    winoff = ((size_t)win * 256 + (y & 15) * 16 + (xc & 15)) * 192;
  }
  #pragma unroll
  for (int j = 0; j < 6; j++) {
    int c = lane + 32 * j;
    u16 o = (c < 180) ? f2bf((v[j] - mean) * inv * g[c] + b[c]) : (u16)0;
    outN[t * 192 + c] = o;
    if (outWin) outWin[winoff + c] = o;
  }
}

// ---------------- generic bf16 WMMA GEMM: out = act(A[M,Kp] @ B[Np,Kp]^T + bias) ----
__global__ __launch_bounds__(256)
void gemm_kernel(const u16* __restrict__ A, const u16* __restrict__ B,
                 const float* __restrict__ bias, const float* __restrict__ resid,
                 float* __restrict__ outF, u16* __restrict__ outB,
                 int N, int Kp, int act, int ldob) {
  int lane = threadIdx.x & 31, wv = threadIdx.x >> 5;
  int wr = wv >> 1, wc = wv & 1;
  int rowBase = blockIdx.y * 128 + wr * 32;
  int colBase = blockIdx.x * 64 + wc * 32;
  int r = lane & 15, k0 = (lane >> 4) * 8;
  const u16* A0 = A + (size_t)(rowBase + r) * Kp + k0;
  const u16* A1 = A0 + (size_t)16 * Kp;
  const u16* B0 = B + (size_t)(colBase + r) * Kp + k0;
  const u16* B1 = B0 + (size_t)16 * Kp;
  v8f acc[2][2] = {};
  for (int k = 0; k < Kp; k += 32) {
    Frag a0, a1, b0, b1;
    a0.h[0] = *(const v8u16*)(A0 + k); a0.h[1] = *(const v8u16*)(A0 + k + 16);
    a1.h[0] = *(const v8u16*)(A1 + k); a1.h[1] = *(const v8u16*)(A1 + k + 16);
    b0.h[0] = *(const v8u16*)(B0 + k); b0.h[1] = *(const v8u16*)(B0 + k + 16);
    b1.h[0] = *(const v8u16*)(B1 + k); b1.h[1] = *(const v8u16*)(B1 + k + 16);
    acc[0][0] = wmma32(a0.v, b0.v, acc[0][0]);
    acc[0][1] = wmma32(a0.v, b1.v, acc[0][1]);
    acc[1][0] = wmma32(a1.v, b0.v, acc[1][0]);
    acc[1][1] = wmma32(a1.v, b1.v, acc[1][1]);
  }
  int lr = (lane >> 4) * 8, lc = lane & 15;
  #pragma unroll
  for (int ti = 0; ti < 2; ti++)
    #pragma unroll
    for (int tj = 0; tj < 2; tj++)
      #pragma unroll
      for (int i = 0; i < 8; i++) {
        int row = rowBase + ti * 16 + lr + i;
        int col = colBase + tj * 16 + lc;
        float v = acc[ti][tj][i];
        if (bias && col < N) v += bias[col];
        if (act == 1) v = 0.5f * v * (1.f + erff(v * 0.70710678f));     // gelu
        else if (act == 2) v = fmaxf(v, 0.f);                            // relu
        if (resid && col < N) v += resid[(size_t)row * N + col];
        if (outF && col < N) outF[(size_t)row * N + col] = v;
        if (outB) outB[(size_t)row * ldob + col] = f2bf(col < N ? v : 0.f);
      }
}

// ---------------- window attention: block = (16 q-rows, head, window) ----------------
__global__ __launch_bounds__(256)
void attn_kernel(const u16* __restrict__ qkv, const float* __restrict__ bias,
                 u16* __restrict__ outb) {
  __shared__ __align__(16) u16 Ksh[256 * 32];     // [key][d]   (B for QK^T)
  __shared__ __align__(16) u16 Vt[32 * 256];      // [d][key]   (B for P*V)
  __shared__ __align__(16) u16 Qsh[16 * 32];
  __shared__ __align__(16) u16 Ptmp[8 * 16 * 32]; // per-wave P slab (A for P*V)
  __shared__ float Opart[8 * 16 * 32];
  __shared__ float rsMax[16][8];
  __shared__ float rsSum[16][8];
  int rb = blockIdx.x, h = blockIdx.y, win = blockIdx.z;
  int tid = threadIdx.x, lane = tid & 31, wv = tid >> 5;
  size_t tb = (size_t)win * 256;
  int qoff = h * 30;
  for (int idx = tid; idx < 256 * 32; idx += 256) {
    int rr = idx >> 5, d = idx & 31;
    u16 kvv = 0, vvv = 0;
    if (d < 30) {
      const u16* p = qkv + (tb + rr) * 576 + qoff + d;
      kvv = p[180]; vvv = p[360];
    }
    Ksh[idx] = kvv;
    Vt[d * 256 + rr] = vvv;
  }
  for (int idx = tid; idx < 16 * 32; idx += 256) {
    int rr = idx >> 5, d = idx & 31;
    u16 qv = 0;
    if (d < 30)
      qv = f2bf(bf2f(qkv[(tb + rb * 16 + rr) * 576 + qoff + d]) * 0.18257418f);
    Qsh[idx] = qv;
  }
  __syncthreads();
  // scores: wave wv owns key tiles 2*wv, 2*wv+1
  Frag qf; qf.v = ldfrag(Qsh, 32, lane);
  Frag k0f, k1f;
  k0f.v = ldfrag(Ksh + (wv * 2) * 16 * 32, 32, lane);
  k1f.v = ldfrag(Ksh + (wv * 2 + 1) * 16 * 32, 32, lane);
  v8f s0 = {}, s1 = {};
  s0 = wmma32(qf.v, k0f.v, s0);
  s1 = wmma32(qf.v, k1f.v, s1);
  const float* bp = bias + (size_t)h * 65536 + (size_t)(rb * 16) * 256;
  int lrb = (lane >> 4) * 8, lc = lane & 15;
  float p0[8], p1[8];
  #pragma unroll
  for (int i = 0; i < 8; i++) {
    int rr = lrb + i;
    float v0 = s0[i] + bp[rr * 256 + wv * 32 + lc];
    float v1 = s1[i] + bp[rr * 256 + wv * 32 + 16 + lc];
    p0[i] = v0; p1[i] = v1;
    float m = fmaxf(v0, v1);
    #pragma unroll
    for (int msk = 8; msk >= 1; msk >>= 1) m = fmaxf(m, __shfl_xor(m, msk, 32));
    if (lc == 0) rsMax[rr][wv] = m;
  }
  __syncthreads();
  #pragma unroll
  for (int i = 0; i < 8; i++) {
    int rr = lrb + i;
    float M = rsMax[rr][0];
    #pragma unroll
    for (int j = 1; j < 8; j++) M = fmaxf(M, rsMax[rr][j]);
    p0[i] = __expf(p0[i] - M);
    p1[i] = __expf(p1[i] - M);
    float sm = p0[i] + p1[i];
    #pragma unroll
    for (int msk = 8; msk >= 1; msk >>= 1) sm += __shfl_xor(sm, msk, 32);
    if (lc == 0) rsSum[rr][wv] = sm;
  }
  __syncthreads();
  #pragma unroll
  for (int i = 0; i < 8; i++) {
    int rr = lrb + i;
    float L = 0.f;
    #pragma unroll
    for (int j = 0; j < 8; j++) L += rsSum[rr][j];
    float inv = 1.f / L;
    Ptmp[wv * 512 + rr * 32 + lc] = f2bf(p0[i] * inv);
    Ptmp[wv * 512 + rr * 32 + 16 + lc] = f2bf(p1[i] * inv);
  }
  __syncthreads();
  // P*V: wave wv owns key slab [wv*32, wv*32+32)
  {
    Frag pa; pa.v = ldfrag(Ptmp + wv * 512, 32, lane);
    Frag vb0, vb1;
    vb0.v = ldfrag(Vt + wv * 32, 256, lane);
    vb1.v = ldfrag(Vt + 16 * 256 + wv * 32, 256, lane);
    v8f o0 = {}, o1 = {};
    o0 = wmma32(pa.v, vb0.v, o0);
    o1 = wmma32(pa.v, vb1.v, o1);
    #pragma unroll
    for (int i = 0; i < 8; i++) {
      int rr = lrb + i;
      Opart[wv * 512 + rr * 32 + lc] = o0[i];
      Opart[wv * 512 + rr * 32 + 16 + lc] = o1[i];
    }
  }
  __syncthreads();
  for (int idx = tid; idx < 16 * 32; idx += 256) {
    int rr = idx >> 5, d = idx & 31;
    float s = 0.f;
    #pragma unroll
    for (int j = 0; j < 8; j++) s += Opart[j * 512 + idx];
    if (d < 30)
      outb[(tb + rb * 16 + rr) * 192 + qoff + d] = f2bf(s);
  }
}

// ---------------- zero pad columns 180..191 of a [M][192] bf16 buffer ----------------
__global__ void padz_kernel(u16* __restrict__ buf) {
  size_t idx = (size_t)blockIdx.x * 256 + threadIdx.x;
  if (idx >= (size_t)MTOK * 12) return;
  size_t t = idx / 12;
  buf[t * 192 + 180 + (idx - t * 12)] = 0;
}

// ---------------- depthwise 3x3 (pad1) + gelu, 90ch, ld 128 ----------------
__global__ void dw1_kernel(const u16* __restrict__ in, const float* __restrict__ w,
                           const float* __restrict__ bias, u16* __restrict__ out) {
  size_t idx = (size_t)blockIdx.x * 256 + threadIdx.x;
  if (idx >= (size_t)MTOK * 128) return;
  int c = (int)(idx & 127);
  size_t t = idx >> 7;
  u16 o = 0;
  if (c < 90) {
    int bb = (int)(t >> 14), rem = (int)(t & 16383);
    int y = rem >> 7, x = rem & 127;
    float acc = bias[c];
    const float* wc = w + c * 9;
    for (int dy = -1; dy <= 1; dy++)
      for (int dx = -1; dx <= 1; dx++) {
        int yy = y + dy, xx = x + dx;
        if (yy >= 0 && yy < HH && xx >= 0 && xx < WW)
          acc += wc[(dy + 1) * 3 + dx + 1] *
                 bf2f(in[(((size_t)bb * 16384 + yy * 128 + xx) << 7) + c]);
      }
    acc = 0.5f * acc * (1.f + erff(acc * 0.70710678f));
    o = f2bf(acc);
  }
  out[idx] = o;
}

// ---------------- depthwise 3x3 dilation2 (pad2), 180ch, bf16 ld192 -> fp32 ld180 ----
__global__ void dw2_kernel(const u16* __restrict__ in, const float* __restrict__ w,
                           const float* __restrict__ bias, float* __restrict__ out) {
  size_t idx = (size_t)blockIdx.x * 256 + threadIdx.x;
  if (idx >= (size_t)MTOK * 180) return;
  size_t t = idx / 180;
  int c = (int)(idx - t * 180);
  int bb = (int)(t >> 14), rem = (int)(t & 16383);
  int y = rem >> 7, x = rem & 127;
  float acc = bias[c];
  const float* wc = w + c * 9;
  for (int dy = -1; dy <= 1; dy++)
    for (int dx = -1; dx <= 1; dx++) {
      int yy = y + 2 * dy, xx = x + 2 * dx;
      if (yy >= 0 && yy < HH && xx >= 0 && xx < WW)
        acc += wc[(dy + 1) * 3 + dx + 1] *
               bf2f(in[((size_t)bb * 16384 + yy * 128 + xx) * 192 + c]);
    }
  out[idx] = acc;
}

// ---------------- depthwise 3x3 (pad1) + relu, 720ch, bf16 ld768 ----------------
__global__ void dw3_kernel(const u16* __restrict__ in, const float* __restrict__ w,
                           const float* __restrict__ bias, u16* __restrict__ out) {
  size_t idx = (size_t)blockIdx.x * 256 + threadIdx.x;
  if (idx >= (size_t)MTOK * 768) return;
  size_t t = idx / 768;
  int c = (int)(idx - t * 768);
  u16 o = 0;
  if (c < 720) {
    int bb = (int)(t >> 14), rem = (int)(t & 16383);
    int y = rem >> 7, x = rem & 127;
    float acc = bias[c];
    const float* wc = w + c * 9;
    for (int dy = -1; dy <= 1; dy++)
      for (int dx = -1; dx <= 1; dx++) {
        int yy = y + dy, xx = x + dx;
        if (yy >= 0 && yy < HH && xx >= 0 && xx < WW)
          acc += wc[(dy + 1) * 3 + dx + 1] *
                 bf2f(in[((size_t)bb * 16384 + yy * 128 + xx) * 768 + c]);
      }
    o = f2bf(fmaxf(acc, 0.f));
  }
  out[idx] = o;
}

// ---------------- global average pool per (b,c) over 16384 pixels ----------------
__global__ __launch_bounds__(256)
void gap_kernel(const float* __restrict__ y3, float* __restrict__ a) {
  int b = blockIdx.x / 180, c = blockIdx.x - b * 180;
  const float* p = y3 + (size_t)b * 16384 * 180 + c;
  float s = 0.f;
  for (int i = threadIdx.x; i < 16384; i += 256) s += p[(size_t)i * 180];
  __shared__ float red[256];
  red[threadIdx.x] = s;
  __syncthreads();
  for (int st = 128; st >= 1; st >>= 1) {
    if (threadIdx.x < st) red[threadIdx.x] += red[threadIdx.x + st];
    __syncthreads();
  }
  if (threadIdx.x == 0) a[blockIdx.x] = red[0] * (1.f / 16384.f);
}

// ---------------- channel attention MLP 180->6 relu ->180 sigmoid ----------------
__global__ void ca_kernel(const float* __restrict__ a, const float* __restrict__ w1,
                          const float* __restrict__ b1, const float* __restrict__ w2,
                          const float* __restrict__ b2, float* __restrict__ a2) {
  int b = blockIdx.x;
  __shared__ float s[6];
  if (threadIdx.x < 6) {
    float acc = b1[threadIdx.x];
    for (int c = 0; c < 180; c++) acc += w1[threadIdx.x * 180 + c] * a[b * 180 + c];
    s[threadIdx.x] = fmaxf(acc, 0.f);
  }
  __syncthreads();
  int c = threadIdx.x;
  if (c < 180) {
    float acc = b2[c];
    #pragma unroll
    for (int j = 0; j < 6; j++) acc += w2[c * 6 + j] * s[j];
    a2[b * 180 + c] = 1.f / (1.f + __expf(-acc));
  }
}

// ---------------- x_new = x + window_reverse(proj) + 0.01 * y3 * a2 ----------------
__global__ void combine_kernel(const float* __restrict__ x, const float* __restrict__ proj,
                               const float* __restrict__ y3, const float* __restrict__ a2,
                               float* __restrict__ xnew) {
  size_t idx = (size_t)blockIdx.x * 256 + threadIdx.x;
  if (idx >= (size_t)MTOK * 180) return;
  size_t t = idx / 180;
  int c = (int)(idx - t * 180);
  int bb = (int)(t >> 14), rem = (int)(t & 16383);
  int y = rem >> 7, xx = rem & 127;
  int win = (bb * 8 + (y >> 4)) * 8 + (xx >> 4);
  size_t wt = (size_t)win * 256 + (y & 15) * 16 + (xx & 15);
  float convx = y3[idx] * a2[bb * 180 + c];
  xnew[idx] = x[idx] + proj[wt * 180 + c] + convx * 0.01f;
}

// ---------------- channel max/mean pool, wave-per-token ----------------
__global__ __launch_bounds__(256)
void pool_kernel(const u16* __restrict__ h, float* __restrict__ pooled) {
  int wv = threadIdx.x >> 5, lane = threadIdx.x & 31;
  size_t t = (size_t)blockIdx.x * 8 + wv;
  const u16* p = h + t * 768;
  float mx = -1e30f, sm = 0.f;
  for (int j = 0; j < 23; j++) {
    int c = lane + 32 * j;
    if (c < 720) { float f = bf2f(p[c]); mx = fmaxf(mx, f); sm += f; }
  }
  #pragma unroll
  for (int m = 16; m >= 1; m >>= 1) {
    mx = fmaxf(mx, __shfl_xor(mx, m, 32));
    sm += __shfl_xor(sm, m, 32);
  }
  if (lane == 0) { pooled[t] = mx; pooled[(size_t)MTOK + t] = sm * (1.f / 720.f); }
}

// ---------------- 7x7 conv over {max,mean} (pad3) + sigmoid ----------------
__global__ void saconv_kernel(const float* __restrict__ pooled, const float* __restrict__ w,
                              float* __restrict__ sa) {
  size_t t = (size_t)blockIdx.x * 256 + threadIdx.x;
  if (t >= (size_t)MTOK) return;
  int bb = (int)(t >> 14), rem = (int)(t & 16383);
  int y = rem >> 7, x = rem & 127;
  float acc = 0.f;
  for (int ch = 0; ch < 2; ch++)
    for (int dy = -3; dy <= 3; dy++)
      for (int dx = -3; dx <= 3; dx++) {
        int yy = y + dy, xx = x + dx;
        if (yy >= 0 && yy < HH && xx >= 0 && xx < WW)
          acc += w[ch * 49 + (dy + 3) * 7 + (dx + 3)] *
                 pooled[(size_t)ch * MTOK + (size_t)bb * 16384 + yy * 128 + xx];
      }
  sa[t] = 1.f / (1.f + __expf(-acc));
}

// ---------------- h *= sa (in place, bf16) ----------------
__global__ void mulsa_kernel(u16* __restrict__ h, const float* __restrict__ sa) {
  size_t idx = (size_t)blockIdx.x * 256 + threadIdx.x;
  if (idx >= (size_t)MTOK * 768) return;
  size_t t = idx / 768;
  h[idx] = f2bf(bf2f(h[idx]) * sa[t]);
}

extern "C" void kernel_launch(void* const* d_in, const int* in_sizes, int n_in,
                              void* d_out, int out_size, void* d_ws, size_t ws_size,
                              hipStream_t stream) {
  (void)in_sizes; (void)n_in; (void)out_size; (void)ws_size;
  const float* x     = (const float*)d_in[0];
  const int*   rpi   = (const int*)d_in[3];
  const float* n1g   = (const float*)d_in[4];
  const float* n1b   = (const float*)d_in[5];
  const float* rpb   = (const float*)d_in[6];
  const float* qkvw  = (const float*)d_in[7];
  const float* qkvb  = (const float*)d_in[8];
  const float* projw = (const float*)d_in[9];
  const float* projb = (const float*)d_in[10];
  const float* cw0   = (const float*)d_in[11];
  const float* cb0   = (const float*)d_in[12];
  const float* cw1   = (const float*)d_in[13];
  const float* cb1   = (const float*)d_in[14];
  const float* cw2   = (const float*)d_in[15];
  const float* cb2   = (const float*)d_in[16];
  const float* cw3   = (const float*)d_in[17];
  const float* cb3   = (const float*)d_in[18];
  const float* caw1  = (const float*)d_in[19];
  const float* cab1  = (const float*)d_in[20];
  const float* caw2  = (const float*)d_in[21];
  const float* cab2  = (const float*)d_in[22];
  const float* n2g   = (const float*)d_in[23];
  const float* n2b   = (const float*)d_in[24];
  const float* fc1w  = (const float*)d_in[25];
  const float* fc1b  = (const float*)d_in[26];
  const float* dww   = (const float*)d_in[27];
  const float* dwb   = (const float*)d_in[28];
  const float* pww   = (const float*)d_in[29];
  const float* pwb   = (const float*)d_in[30];
  const float* saw   = (const float*)d_in[31];
  const float* fc2w  = (const float*)d_in[32];
  const float* fc2b  = (const float*)d_in[33];
  float* outp = (float*)d_out;

  char* wsb = (char*)d_ws;
  size_t off = 0;
  auto alloc = [&](size_t bytes) {
    void* p = wsb + off;
    off += (bytes + 255) & ~(size_t)255;
    return p;
  };
  const size_t M = MTOK;
  u16* qkvw_b  = (u16*)alloc(576 * 192 * 2);
  u16* projw_b = (u16*)alloc(192 * 192 * 2);
  u16* cab0w_b = (u16*)alloc(128 * 192 * 2);
  u16* cab2w_b = (u16*)alloc(192 * 128 * 2);
  u16* fc1w_b  = (u16*)alloc(768 * 192 * 2);
  u16* pww_b   = (u16*)alloc(768 * 768 * 2);
  u16* fc2w_b  = (u16*)alloc(192 * 768 * 2);
  float* biasA = (float*)alloc((size_t)6 * 65536 * 4);
  u16* xnwin   = (u16*)alloc(M * 192 * 2);
  u16* xnb     = (u16*)alloc(M * 192 * 2);
  u16* qkvb16  = (u16*)alloc(M * 576 * 2);
  u16* attnout = (u16*)alloc(M * 192 * 2);
  float* projout = (float*)alloc(M * 180 * 4);
  u16* y0b     = (u16*)alloc(M * 128 * 2);
  u16* y1b     = (u16*)alloc(M * 128 * 2);
  u16* y2b     = (u16*)alloc(M * 192 * 2);
  float* y3f   = (float*)alloc(M * 180 * 4);
  float* gapa  = (float*)alloc(4 * 180 * 4);
  float* caa2  = (float*)alloc(4 * 180 * 4);
  float* xnew  = (float*)alloc(M * 180 * 4);
  u16* xn2b    = (u16*)alloc(M * 192 * 2);
  u16* h1b     = (u16*)alloc(M * 768 * 2);
  u16* h2b     = (u16*)alloc(M * 768 * 2);
  u16* h3b     = (u16*)alloc(M * 768 * 2);
  float* pooled = (float*)alloc(2 * M * 4);
  float* sabuf  = (float*)alloc(M * 4);

  auto cvt = [&](const float* w, u16* o, int N, int K, int Np, int Kp) {
    int n = Np * Kp;
    cvt_w_kernel<<<(n + 255) / 256, 256, 0, stream>>>(w, o, N, K, Np, Kp);
  };
  cvt(qkvw, qkvw_b, 540, 180, 576, 192);
  cvt(projw, projw_b, 180, 180, 192, 192);
  cvt(cw0, cab0w_b, 90, 180, 128, 192);
  cvt(cw2, cab2w_b, 180, 90, 192, 128);
  cvt(fc1w, fc1w_b, 720, 180, 768, 192);
  cvt(pww, pww_b, 720, 720, 768, 768);
  cvt(fc2w, fc2w_b, 180, 720, 192, 768);

  build_bias_kernel<<<(6 * 65536) / 256, 256, 0, stream>>>(rpi, rpb, biasA);

  // LN1 -> bf16 (normal order + window order)
  ln_kernel<<<M / 8, 256, 0, stream>>>(x, n1g, n1b, xnb, xnwin);

  padz_kernel<<<(int)((M * 12) / 256), 256, 0, stream>>>(attnout);

  // qkv GEMM (window order)
  gemm_kernel<<<dim3(576 / 64, M / 128), 256, 0, stream>>>(
      xnwin, qkvw_b, qkvb, nullptr, nullptr, qkvb16, 540, 192, 0, 576);

  // window attention
  attn_kernel<<<dim3(16, 6, 256), 256, 0, stream>>>(qkvb16, biasA, attnout);

  // proj GEMM (window order, fp32 out)
  gemm_kernel<<<dim3(192 / 64, M / 128), 256, 0, stream>>>(
      attnout, projw_b, projb, nullptr, projout, nullptr, 180, 192, 0, 0);

  // CAB branch
  gemm_kernel<<<dim3(128 / 64, M / 128), 256, 0, stream>>>(
      xnb, cab0w_b, cb0, nullptr, nullptr, y0b, 90, 192, 0, 128);
  dw1_kernel<<<(int)((M * 128) / 256), 256, 0, stream>>>(y0b, cw1, cb1, y1b);
  gemm_kernel<<<dim3(192 / 64, M / 128), 256, 0, stream>>>(
      y1b, cab2w_b, cb2, nullptr, nullptr, y2b, 180, 128, 0, 192);
  dw2_kernel<<<(int)((M * 180) / 256), 256, 0, stream>>>(y2b, cw3, cb3, y3f);
  gap_kernel<<<720, 256, 0, stream>>>(y3f, gapa);
  ca_kernel<<<4, 192, 0, stream>>>(gapa, caw1, cab1, caw2, cab2, caa2);

  combine_kernel<<<(int)((M * 180) / 256), 256, 0, stream>>>(x, projout, y3f, caa2, xnew);

  // FFN
  ln_kernel<<<M / 8, 256, 0, stream>>>(xnew, n2g, n2b, xn2b, nullptr);
  gemm_kernel<<<dim3(768 / 64, M / 128), 256, 0, stream>>>(
      xn2b, fc1w_b, fc1b, nullptr, nullptr, h1b, 720, 192, 1 /*gelu*/, 768);
  dw3_kernel<<<(int)((M * 768) / 256), 256, 0, stream>>>(h1b, dww, dwb, h2b);
  gemm_kernel<<<dim3(768 / 64, M / 128), 256, 0, stream>>>(
      h2b, pww_b, pwb, nullptr, nullptr, h3b, 720, 768, 2 /*relu*/, 768);
  pool_kernel<<<M / 8, 256, 0, stream>>>(h3b, pooled);
  saconv_kernel<<<(int)(M / 256), 256, 0, stream>>>(pooled, saw, sabuf);
  mulsa_kernel<<<(int)((M * 768) / 256), 256, 0, stream>>>(h3b, sabuf);

  // fc2 GEMM with fused residual -> final output
  gemm_kernel<<<dim3(192 / 64, M / 128), 256, 0, stream>>>(
      h3b, fc2w_b, fc2b, xnew, outp, nullptr, 180, 768, 0, 0);
}
